// CausalSelfAttention_12335146074592
// MI455X (gfx1250) — matrix operand order repaired and checked by reading, not compiled, not run on hardware
//
#include <hip/hip_runtime.h>

// ---------------------------------------------------------------- types
typedef __bf16 bf16;
typedef __attribute__((ext_vector_type(16))) bf16  v16bf;
typedef __attribute__((ext_vector_type(8)))  bf16  v8bf;
typedef __attribute__((ext_vector_type(8)))  float v8f;

#define T_SEQ 2048
#define C_EMB 1024
#define HQ    16
#define HKV   8
#define DHEAD 64
#define NCOL  2048   // wmma columns of qkv (gates handled separately)

__device__ inline bf16 f2bf(float f) {
  unsigned u = __float_as_uint(f);
  unsigned r = (u + 0x7fffu + ((u >> 16) & 1u)) >> 16;
  unsigned short s = (unsigned short)r;
  return __builtin_bit_cast(bf16, s);
}

__device__ inline v8f vzero() {
  v8f c;
#pragma unroll
  for (int i = 0; i < 8; ++i) c[i] = 0.0f;
  return c;
}

__device__ inline v8f wmma_bf16(v16bf a, v16bf b, v8f c) {
  return __builtin_amdgcn_wmma_f32_16x16x32_bf16(false, a, false, b,
                                                 (short)0, c, false, false);
}

// A fragment 16x32 (MxK), row-major source, per ISA layout:
// lanes 0-15: M=lane, K = {k0..k0+7} U {k0+16..k0+23}
// lanes16-31: M=lane-16, K = {k0+8..k0+15} U {k0+24..k0+31}
__device__ inline v16bf load_frag_A(const bf16* __restrict__ base, int ld,
                                    int row0, int k0) {
  int lane  = threadIdx.x & 31;
  int m     = lane & 15;
  int khalf = (lane >> 4) << 3;  // 0 or 8
  const bf16* p = base + (size_t)(row0 + m) * ld + k0 + khalf;
  v8bf lo = *(const v8bf*)p;
  v8bf hi = *(const v8bf*)(p + 16);
  v16bf r;
#pragma unroll
  for (int i = 0; i < 8; ++i) { r[i] = lo[i]; r[i + 8] = hi[i]; }
  return r;
}

// B fragment 32x16 (KxN) read from a transposed [n][k] buffer:
// lanes 0-15: N=lane, K = {k0..k0+7} U {k0+16..k0+23} ; lanes16-31: N=lane-16, K = {k0+8..k0+15} U {k0+24..k0+31}
__device__ inline v16bf load_frag_BT(const bf16* __restrict__ baseT, int ld,
                                     int col0, int k0) {
  int lane = threadIdx.x & 31;
  int n    = lane & 15;
  int kb   = (lane >> 4) << 3;  // 0 or 8
  const bf16* p = baseT + (size_t)(col0 + n) * ld + k0 + kb;
  v8bf lo = *(const v8bf*)p;
  v8bf hi = *(const v8bf*)(p + 16);
  v16bf r;
#pragma unroll
  for (int i = 0; i < 8; ++i) { r[i] = lo[i]; r[i + 8] = hi[i]; }
  return r;
}

// ---------------------------------------------------------------- prep
__global__ void prep_convert(const float* __restrict__ x,
                             const float* __restrict__ w_attn,
                             const float* __restrict__ w_proj,
                             bf16* __restrict__ xb, bf16* __restrict__ waT,
                             bf16* __restrict__ wpT) {
  size_t i = (size_t)blockIdx.x * blockDim.x + threadIdx.x;
  const size_t NX = (size_t)T_SEQ * C_EMB;
  const size_t NW = (size_t)NCOL * C_EMB;
  const size_t NP = (size_t)C_EMB * C_EMB;
  if (i < NX) {
    xb[i] = f2bf(x[i]);
  } else if (i < NX + NW) {
    size_t j = i - NX;
    int n = (int)(j >> 10), k = (int)(j & 1023);
    waT[j] = f2bf(w_attn[(size_t)k * (NCOL + HKV) + n]);
  } else if (i < NX + NW + NP) {
    size_t j = i - NX - NW;
    int n = (int)(j >> 10), k = (int)(j & 1023);
    wpT[j] = f2bf(w_proj[(size_t)k * C_EMB + n]);
  }
}

// ---------------------------------------------------------------- bf16 GEMM
// C[M,N] (f32) = A[M,K](bf16 row-major) * B (bf16, stored transposed BT[N][K])
// block = 128 threads (4 waves in 2x2), block tile 128x128, wave tile 64x64.
__global__ __launch_bounds__(128) void gemm_bf16(const bf16* __restrict__ A,
                                                 const bf16* __restrict__ BT,
                                                 float* __restrict__ C, int M,
                                                 int N, int K) {
  int wave = threadIdx.x >> 5;
  int lane = threadIdx.x & 31;
  int row0 = blockIdx.y * 128 + (wave >> 1) * 64;
  int col0 = blockIdx.x * 128 + (wave & 1) * 64;

  v8f acc[4][4];
#pragma unroll
  for (int i = 0; i < 4; ++i)
#pragma unroll
    for (int j = 0; j < 4; ++j) acc[i][j] = vzero();

  for (int k0 = 0; k0 < K; k0 += 32) {
    v16bf af[4], bfm[4];
#pragma unroll
    for (int i = 0; i < 4; ++i) af[i] = load_frag_A(A, K, row0 + 16 * i, k0);
#pragma unroll
    for (int j = 0; j < 4; ++j) bfm[j] = load_frag_BT(BT, K, col0 + 16 * j, k0);
#pragma unroll
    for (int i = 0; i < 4; ++i)
#pragma unroll
      for (int j = 0; j < 4; ++j) acc[i][j] = wmma_bf16(af[i], bfm[j], acc[i][j]);
  }

  int half8 = (lane >> 4) << 3;
  int cn    = lane & 15;
#pragma unroll
  for (int i = 0; i < 4; ++i)
#pragma unroll
    for (int j = 0; j < 4; ++j)
#pragma unroll
      for (int r = 0; r < 8; ++r)
        C[(size_t)(row0 + 16 * i + r + half8) * N + col0 + 16 * j + cn] =
            acc[i][j][r];
}

// ---------------------------------------------------------------- gates (fp32)
__global__ void gates_kernel(const float* __restrict__ x,
                             const float* __restrict__ w_attn,
                             float* __restrict__ log_g) {
  int idx = blockIdx.x * blockDim.x + threadIdx.x;
  if (idx >= T_SEQ * HKV) return;
  int t = idx >> 3, gd = idx & 7;
  const float* xr = x + (size_t)t * C_EMB;
  const float* wc = w_attn + NCOL + gd;  // column, stride 2056
  float s = 0.f;
  for (int k = 0; k < C_EMB; ++k) s += xr[k] * wc[(size_t)k * (NCOL + HKV)];
  float z = 6.906768f + s;
  float lg = (z >= 0.f) ? -log1pf(__expf(-z)) : z - log1pf(__expf(z));
  log_g[idx] = lg;  // [t][g]
}

// cumsum over T per gate head: 1 wave per head
__global__ void scan_kernel(const float* __restrict__ log_g,
                            float* __restrict__ logG) {
  int gd = blockIdx.x;
  int lane = threadIdx.x;
  int base = lane * 64;
  float sum = 0.f;
  for (int i = 0; i < 64; ++i) sum += log_g[(size_t)(base + i) * HKV + gd];
  float ex = sum;
#pragma unroll
  for (int off = 1; off < 32; off <<= 1) {
    float nv = __shfl_up(ex, off, 32);
    if (lane >= off) ex += nv;
  }
  ex -= sum;  // exclusive prefix
  float run = ex;
  for (int i = 0; i < 64; ++i) {
    run += log_g[(size_t)(base + i) * HKV + gd];
    logG[(size_t)gd * T_SEQ + base + i] = run;
  }
}

// ---------------------------------------------------------------- rope / pack
__global__ void rope_pack(const float* __restrict__ qkvg,
                          bf16* __restrict__ qb, bf16* __restrict__ kb,
                          bf16* __restrict__ vT) {
  const int NQ = T_SEQ * HQ * 32;   // q rotation pairs
  const int NK = T_SEQ * HKV * 32;  // k rotation pairs
  const int NV = T_SEQ * HKV * 64;  // v elements
  int i = blockIdx.x * blockDim.x + threadIdx.x;
  if (i < NQ) {
    int j = i & 31, h = (i >> 5) & 15, t = i >> 9;
    const float* p = qkvg + (size_t)t * NCOL + h * DHEAD + 2 * j;
    float ang = (float)t * 6.28318530718f * __powf(2048.f, -(float)j / 32.f);
    float sn, cs; __sincosf(ang, &sn, &cs);
    float x1 = p[0], x2 = p[1];
    bf16* q = qb + (size_t)t * (HQ * DHEAD) + h * DHEAD + 2 * j;
    q[0] = f2bf(x1 * cs - x2 * sn);
    q[1] = f2bf(x2 * cs + x1 * sn);
  } else if (i < NQ + NK) {
    int m = i - NQ;
    int j = m & 31, g = (m >> 5) & 7, t = m >> 8;
    const float* p = qkvg + (size_t)t * NCOL + (HQ * DHEAD) + g * DHEAD + 2 * j;
    float ang = (float)t * 6.28318530718f * __powf(2048.f, -(float)j / 32.f);
    float sn, cs; __sincosf(ang, &sn, &cs);
    float x1 = p[0], x2 = p[1];
    bf16* k = kb + (size_t)t * (HKV * DHEAD) + g * DHEAD + 2 * j;
    k[0] = f2bf(x1 * cs - x2 * sn);
    k[1] = f2bf(x2 * cs + x1 * sn);
  } else if (i < NQ + NK + NV) {
    int m = i - NQ - NK;
    int d = m & 63, g = (m >> 6) & 7, t = m >> 9;
    float v = qkvg[(size_t)t * NCOL + (HQ + HKV) * DHEAD + g * DHEAD + d];
    vT[((size_t)g * DHEAD + d) * T_SEQ + t] = f2bf(v);  // [g][d][t]
  }
}

// ---------------------------------------------------------------- attention
// grid = (16 heads, T/128); block = 256 (8 waves); wave handles 16 query rows.
// K/V tiles (64 keys) are staged once per block into LDS and shared by all
// 8 waves (8x less L2 traffic); next tile is prefetched during compute.
__global__ __launch_bounds__(256) void attn_kernel(
    const bf16* __restrict__ qb, const bf16* __restrict__ kb,
    const bf16* __restrict__ vT, const float* __restrict__ logG,
    const float* __restrict__ lnw, bf16* __restrict__ yb) {
  __shared__ __align__(16) bf16 klds[64 * 64];      // [key][d]
  __shared__ __align__(16) bf16 vlds[64 * 64];      // [d][key]
  __shared__ __align__(16) bf16 a_lds[8][16 * 64];  // per-wave prob tile

  int wave = threadIdx.x >> 5;
  int lane = threadIdx.x & 31;
  int h = blockIdx.x;
  int g = h >> 1;
  int t0blk = blockIdx.y * 128;
  int t0 = t0blk + wave * 16;
  int half8 = (lane >> 4) << 3;
  int cn = lane & 15;
  const float scale = 0.125f;  // 1/sqrt(64)

  // q fragments for this 16-row strip (K chunks 0 and 32 of head dim)
  v16bf qf[2];
#pragma unroll
  for (int kk = 0; kk < 2; ++kk)
    qf[kk] = load_frag_A(qb + h * DHEAD, HQ * DHEAD, t0, kk * 32);

  // per-row logG[t]
  float lgt[8];
#pragma unroll
  for (int r = 0; r < 8; ++r)
    lgt[r] = logG[(size_t)g * T_SEQ + t0 + r + half8];

  float mrow[8], den[8];
  v8f acc[4];
#pragma unroll
  for (int r = 0; r < 8; ++r) { mrow[r] = -__builtin_inff(); den[r] = 0.f; }
#pragma unroll
  for (int dt = 0; dt < 4; ++dt) acc[dt] = vzero();

  bf16* al = a_lds[wave];
  const int smax = t0blk + 128;  // block-uniform tile loop bound

  for (int s0 = 0; s0 < smax; s0 += 64) {
    // ---- cooperative K/V tile staging (256 threads, 32B each per tile)
    {
      int tid = threadIdx.x;
      int row = tid >> 2;            // 0..63
      int seg = (tid & 3) << 4;      // 0,16,32,48
      const bf16* ksrc = kb + (size_t)(s0 + row) * (HKV * DHEAD) + g * DHEAD + seg;
      *(v8bf*)(klds + row * 64 + seg)     = *(const v8bf*)ksrc;
      *(v8bf*)(klds + row * 64 + seg + 8) = *(const v8bf*)(ksrc + 8);
      const bf16* vsrc = vT + ((size_t)g * DHEAD + row) * T_SEQ + s0 + seg;
      *(v8bf*)(vlds + row * 64 + seg)     = *(const v8bf*)vsrc;
      *(v8bf*)(vlds + row * 64 + seg + 8) = *(const v8bf*)(vsrc + 8);
      if (s0 + 64 < smax) {  // prefetch next tile into caches
        __builtin_prefetch(ksrc + (size_t)64 * (HKV * DHEAD), 0, 1);
        __builtin_prefetch(vsrc + 64, 0, 1);
      }
    }
    __syncthreads();

    if (s0 <= t0 + 15) {  // tiles fully above the diagonal are skipped
      // ---- S = q @ k^T for a 16x64 tile (k from LDS)
      v8f sfr[4];
#pragma unroll
      for (int ct = 0; ct < 4; ++ct) {
        v8f c = vzero();
#pragma unroll
        for (int kk = 0; kk < 2; ++kk) {
          v16bf b = load_frag_BT(klds, DHEAD, ct * 16, kk * 32);
          c = wmma_bf16(qf[kk], b, c);
        }
        sfr[ct] = c;
      }
      // ---- logits (in place) + tile row max
      float rm[8];
#pragma unroll
      for (int r = 0; r < 8; ++r) rm[r] = -__builtin_inff();
#pragma unroll
      for (int ct = 0; ct < 4; ++ct) {
        int s_idx = s0 + ct * 16 + cn;
        float lgs = logG[(size_t)g * T_SEQ + s_idx];
#pragma unroll
        for (int r = 0; r < 8; ++r) {
          int t_idx = t0 + r + half8;
          float sv = sfr[ct][r] * scale;
          float lg = 2.f * __logf(fabsf(sv) + 1e-30f) + lgt[r] - lgs;
          lg = (s_idx <= t_idx) ? lg : -__builtin_inff();
          sfr[ct][r] = lg;
          rm[r] = fmaxf(rm[r], lg);
        }
      }
#pragma unroll
      for (int off = 8; off >= 1; off >>= 1)
#pragma unroll
        for (int r = 0; r < 8; ++r)
          rm[r] = fmaxf(rm[r], __shfl_xor(rm[r], off, 32));

      float scl[8], rs[8];
#pragma unroll
      for (int r = 0; r < 8; ++r) {
        float mn = fmaxf(mrow[r], rm[r]);
        scl[r] = __expf(mrow[r] - mn);
        mrow[r] = mn;
        rs[r] = 0.f;
      }
      // ---- probs, row sums, stage into per-wave LDS in (row, col) order
#pragma unroll
      for (int ct = 0; ct < 4; ++ct)
#pragma unroll
        for (int r = 0; r < 8; ++r) {
          float lg = sfr[ct][r];
          float a = (lg == -__builtin_inff()) ? 0.f : __expf(lg - mrow[r]);
          rs[r] += a;
          al[(r + half8) * 64 + ct * 16 + cn] = f2bf(a);
        }
#pragma unroll
      for (int off = 8; off >= 1; off >>= 1)
#pragma unroll
        for (int r = 0; r < 8; ++r) rs[r] += __shfl_xor(rs[r], off, 32);
#pragma unroll
      for (int r = 0; r < 8; ++r) den[r] = den[r] * scl[r] + rs[r];
#pragma unroll
      for (int dt = 0; dt < 4; ++dt)
#pragma unroll
        for (int r = 0; r < 8; ++r) acc[dt][r] *= scl[r];

      __builtin_amdgcn_wave_barrier();
      asm volatile("s_wait_dscnt 0x0" ::: "memory");

      // ---- num += A @ V  (A from per-wave LDS, V tile from shared LDS)
      v16bf af[2];
#pragma unroll
      for (int kk = 0; kk < 2; ++kk) af[kk] = load_frag_A(al, 64, 0, kk * 32);
#pragma unroll
      for (int dt = 0; dt < 4; ++dt)
#pragma unroll
        for (int kk = 0; kk < 2; ++kk) {
          v16bf b = load_frag_BT(vlds, DHEAD, dt * 16, kk * 32);
          acc[dt] = wmma_bf16(af[kk], b, acc[dt]);
        }
      __builtin_amdgcn_wave_barrier();
    }
    __syncthreads();
  }

  // ---- y = num/den, then LayerNorm over head dim (64) with ln_w
  float inv[8];
#pragma unroll
  for (int r = 0; r < 8; ++r) inv[r] = 1.f / (den[r] + 1e-6f);
  v8f yv[4];
#pragma unroll
  for (int dt = 0; dt < 4; ++dt)
#pragma unroll
    for (int r = 0; r < 8; ++r) yv[dt][r] = acc[dt][r] * inv[r];

  float mu[8];
#pragma unroll
  for (int r = 0; r < 8; ++r) mu[r] = 0.f;
#pragma unroll
  for (int dt = 0; dt < 4; ++dt)
#pragma unroll
    for (int r = 0; r < 8; ++r) mu[r] += yv[dt][r];
#pragma unroll
  for (int off = 8; off >= 1; off >>= 1)
#pragma unroll
    for (int r = 0; r < 8; ++r) mu[r] += __shfl_xor(mu[r], off, 32);
#pragma unroll
  for (int r = 0; r < 8; ++r) mu[r] *= (1.f / 64.f);

  float var[8];
#pragma unroll
  for (int r = 0; r < 8; ++r) var[r] = 0.f;
#pragma unroll
  for (int dt = 0; dt < 4; ++dt)
#pragma unroll
    for (int r = 0; r < 8; ++r) {
      float d = yv[dt][r] - mu[r];
      var[r] += d * d;
    }
#pragma unroll
  for (int off = 8; off >= 1; off >>= 1)
#pragma unroll
    for (int r = 0; r < 8; ++r) var[r] += __shfl_xor(var[r], off, 32);
#pragma unroll
  for (int r = 0; r < 8; ++r) var[r] = rsqrtf(var[r] * (1.f / 64.f) + 1e-9f);

#pragma unroll
  for (int dt = 0; dt < 4; ++dt) {
    float w = lnw[dt * 16 + cn];
#pragma unroll
    for (int r = 0; r < 8; ++r) {
      float o = (yv[dt][r] - mu[r]) * var[r] * w;
      yb[(size_t)(t0 + r + half8) * (HQ * DHEAD) + h * DHEAD + dt * 16 + cn] =
          f2bf(o);
    }
  }
}

// ---------------------------------------------------------------- launch
extern "C" void kernel_launch(void* const* d_in, const int* in_sizes, int n_in,
                              void* d_out, int out_size, void* d_ws,
                              size_t ws_size, hipStream_t stream) {
  const float* x      = (const float*)d_in[0];
  const float* w_attn = (const float*)d_in[1];
  const float* w_proj = (const float*)d_in[2];
  const float* ln_w   = (const float*)d_in[3];
  float* out = (float*)d_out;

  char* ws = (char*)d_ws;
  bf16* xb    = (bf16*)(ws + 0);                   // 4 MB
  bf16* waT   = (bf16*)(ws + 4194304);             // 4 MB
  bf16* wpT   = (bf16*)(ws + 8388608);             // 2 MB
  float* qkvg = (float*)(ws + 10485760);           // 16 MB  [T][2048]
  float* logg = (float*)(ws + 27262976);           // 64 KB  [T][8]
  float* logG = (float*)(ws + 27328512);           // 64 KB  [8][T]
  bf16* qb    = (bf16*)(ws + 27394048);            // 4 MB   [T][16][64]
  bf16* kb    = (bf16*)(ws + 31588352);            // 2 MB   [T][8][64]
  bf16* vT    = (bf16*)(ws + 33685504);            // 2 MB   [8][64][T]
  bf16* yb    = (bf16*)(ws + 35782656);            // 4 MB   [T][1024]

  // 1) convert / transpose weights & activations to bf16
  {
    size_t total = (size_t)T_SEQ * C_EMB + (size_t)NCOL * C_EMB +
                   (size_t)C_EMB * C_EMB;
    prep_convert<<<(unsigned)((total + 255) / 256), 256, 0, stream>>>(
        x, w_attn, w_proj, xb, waT, wpT);
  }
  // 2) QKV projection (2048 wmma columns)
  gemm_bf16<<<dim3(NCOL / 128, T_SEQ / 128), 128, 0, stream>>>(
      xb, waT, qkvg, T_SEQ, NCOL, C_EMB);
  // 3) gate columns in fp32 + log-sigmoid
  gates_kernel<<<(T_SEQ * HKV + 255) / 256, 256, 0, stream>>>(x, w_attn, logg);
  // 4) cumulative gate sums (1 wave per kv head)
  scan_kernel<<<HKV, 32, 0, stream>>>(logg, logG);
  // 5) RoPE + repack q/k/v
  {
    int total = T_SEQ * HQ * 32 + T_SEQ * HKV * 32 + T_SEQ * HKV * 64;
    rope_pack<<<(total + 255) / 256, 256, 0, stream>>>(qkvg, qb, kb, vT);
  }
  // 6) gated power attention + LayerNorm
  attn_kernel<<<dim3(HQ, T_SEQ / 128), 256, 0, stream>>>(qb, kb, vT, logG,
                                                         ln_w, yb);
  // 7) output projection
  gemm_bf16<<<dim3(C_EMB / 128, T_SEQ / 128), 128, 0, stream>>>(
      yb, wpT, out, T_SEQ, C_EMB, C_EMB);
}
